// TFS_73005854098209
// MI455X (gfx1250) — matrix operand order repaired and checked by reading, not compiled, not run on hardware
//
#include <hip/hip_runtime.h>
#include <hip/hip_bf16.h>

// ---------------------------------------------------------------------------
// MI455X (gfx1250) implementation.
// - All large GEMMs route through V_WMMA_F32_16X16X32_F16 (f16 in, f32 acc).
// - GEMM: block = 64M x 64N; each wave owns 4 stacked 16x16 tiles so one B
//   fragment feeds 4 WMMAs (register reuse); the shared B tile is staged into
//   LDS by the Tensor Data Mover (tensor_load_to_lds, double-buffered,
//   s_wait_tensorcnt + barrier), overlapping DMA with WMMA compute.
// - LSTM input projections batched into one [65536x1024x256] GEMM per layer;
//   only h@Whh^T stays per-timestep.
// ---------------------------------------------------------------------------

typedef _Float16 f16;
typedef _Float16 v16h __attribute__((ext_vector_type(16)));
typedef _Float16 v8h  __attribute__((ext_vector_type(8)));
typedef float    v8f  __attribute__((ext_vector_type(8)));
typedef unsigned int v4u __attribute__((ext_vector_type(4)));
typedef int      v4i __attribute__((ext_vector_type(4)));
typedef int      v8i __attribute__((ext_vector_type(8)));

union V16 { v16h v; v8h h[2]; };

static constexpr int cB  = 1024;
static constexpr int cT  = 64;
static constexpr int cIN = 14;
static constexpr int cE  = 256;
static constexpr int cNH = 8;
static constexpr int cDH = 32;
static constexpr int cLH = 512;
static constexpr int cR  = cB * cT;   // 65536 rows
static constexpr int cG  = 4 * cE;    // 1024 LSTM gates

#if defined(__has_builtin)
#if __has_builtin(__builtin_amdgcn_tensor_load_to_lds)
#define HAVE_TDM 1
#endif
#endif
#ifndef HAVE_TDM
#define HAVE_TDM 0
#endif

#if HAVE_TDM
// Issue a TDM load of a 64-row x 64-element (f16) 2D tile into LDS.
// D# per CDNA5 ISA ch.8: group0 = {count, lds_addr, global_addr, type=2};
// group1 = {data_size=2B, tensor dims (large), tile 64x64, dim0 stride}.
__device__ __forceinline__ void tdm_load_tile64(unsigned lds_off, const void* gptr,
                                                unsigned ld_elems)
{
    unsigned long long ga = (unsigned long long)gptr;
    v4u g0;
    g0[0] = 1u;                                   // count=1 valid descriptor
    g0[1] = lds_off;                              // lds_addr (bytes)
    g0[2] = (unsigned)(ga & 0xffffffffu);         // global_addr[31:0]
    g0[3] = (unsigned)((ga >> 32) & 0x01ffffffu)  // global_addr[56:32]
          | (2u << 30);                           // type=2 ("image")
    v8i g1;
    g1[0] = 0x00010000;                           // wg_mask=0, data_size=1 (2B)
    g1[1] = 0;                                    // abar=0, tensor_dim0 lo16=0
    g1[2] = 0x4000;                               // tensor_dim0=0x4000_0000, td1 lo=0
    g1[3] = (64 << 16) | 0x4000;                  // td1 hi (0x4000_0000), tile_dim0=64
    g1[4] = 64;                                   // tile_dim1=64, tile_dim2=0
    g1[5] = (int)ld_elems;                        // tensor_dim0_stride[31:0]
    g1[6] = 0;                                    // stride hi / td1_stride lo
    g1[7] = 0;                                    // td1_stride hi
    v4i z4 = {0, 0, 0, 0};
#if __clang_major__ >= 23
    v8i z8 = {0, 0, 0, 0, 0, 0, 0, 0};
    __builtin_amdgcn_tensor_load_to_lds(g0, g1, z4, z4, z8, 0);
#else
    __builtin_amdgcn_tensor_load_to_lds(g0, g1, z4, z4, 0);
#endif
}
#endif

// ---------------------------------------------------------------------------
// WMMA GEMM:  C[M,N] = A[M,K](f16) * Bt[N,K](f16)^T + bias[N]
// Block: 64(M) x 64(N); 4 waves, wave w owns N-cols [w*16, w*16+16) and 4
// stacked M-subtiles -> 4 WMMAs per B fragment. B tile (64x64 f16 = 8KB)
// double-buffered in LDS via TDM; A streamed from global (pipelined b128).
// Requires: M%64==0, N%64==0, K%64==0.
// ---------------------------------------------------------------------------
__global__ void __launch_bounds__(128)
gemm_wmma_f16(const f16* __restrict__ A, int lda,
              const f16* __restrict__ Bt, int ldb,
              const float* __restrict__ bias,
              float* __restrict__ Cf, f16* __restrict__ Ch,
              int ldc, int K, int relu)
{
    __shared__ __align__(64) f16 ldsB[2][64 * 64];   // 16 KB double buffer

    const int lane = threadIdx.x & 31;
    const int wave = threadIdx.x >> 5;
    const int row0 = blockIdx.y << 6;
    const int col0 = (blockIdx.x << 6) + (wave << 4);
    const int mn   = lane & 15;
    const int hlf  = lane >> 4;

    const f16* ap = A + (size_t)(row0 + mn) * lda + 16 * hlf;
    const size_t arow = (size_t)16 * lda;

    v8f acc0 = {}, acc1 = {}, acc2 = {}, acc3 = {};

#if HAVE_TDM
    const f16* btile = Bt + (size_t)(blockIdx.x << 6) * ldb;
    if (wave == 0)
        tdm_load_tile64((unsigned)(unsigned long long)&ldsB[0][0], btile, (unsigned)ldb);
    const int nkc = K >> 6;
    for (int kc = 0; kc < nkc; ++kc) {
        __builtin_amdgcn_s_wait_tensorcnt(0);   // chunk kc resident (wave0; others no-op)
        __syncthreads();
        if (wave == 0 && kc + 1 < nkc)          // prefetch chunk kc+1 into other buffer
            tdm_load_tile64((unsigned)(unsigned long long)&ldsB[(kc + 1) & 1][0],
                            btile + (size_t)(kc + 1) * 64, (unsigned)ldb);
        const f16* bp = &ldsB[kc & 1][((wave << 4) + mn) * 64 + 16 * hlf];
        const f16* ak = ap + (size_t)kc * 64;
#pragma unroll
        for (int ks = 0; ks < 64; ks += 32) {
            V16 b;
            b.v = *(const v16h*)(bp + ks);      // ds_load from staged tile
            V16 a0, a1, a2, a3;
            a0.h[0] = *(const v8h*)(ak + ks);
            a0.h[1] = *(const v8h*)(ak + ks + 8);
            a1.h[0] = *(const v8h*)(ak + arow + ks);
            a1.h[1] = *(const v8h*)(ak + arow + ks + 8);
            a2.h[0] = *(const v8h*)(ak + 2 * arow + ks);
            a2.h[1] = *(const v8h*)(ak + 2 * arow + ks + 8);
            a3.h[0] = *(const v8h*)(ak + 3 * arow + ks);
            a3.h[1] = *(const v8h*)(ak + 3 * arow + ks + 8);
            acc0 = __builtin_amdgcn_wmma_f32_16x16x32_f16(false, a0.v, false, b.v, (short)0, acc0, false, false);
            acc1 = __builtin_amdgcn_wmma_f32_16x16x32_f16(false, a1.v, false, b.v, (short)0, acc1, false, false);
            acc2 = __builtin_amdgcn_wmma_f32_16x16x32_f16(false, a2.v, false, b.v, (short)0, acc2, false, false);
            acc3 = __builtin_amdgcn_wmma_f32_16x16x32_f16(false, a3.v, false, b.v, (short)0, acc3, false, false);
        }
    }
#else
    const f16* bp = Bt + (size_t)(col0 + mn) * ldb + 16 * hlf;
    for (int k = 0; k < K; k += 32) {
        V16 b;
        b.v = *(const v16h*)(bp + k);
        V16 a0, a1, a2, a3;
        a0.h[0] = *(const v8h*)(ap + k);
        a0.h[1] = *(const v8h*)(ap + k + 8);
        a1.h[0] = *(const v8h*)(ap + arow + k);
        a1.h[1] = *(const v8h*)(ap + arow + k + 8);
        a2.h[0] = *(const v8h*)(ap + 2 * arow + k);
        a2.h[1] = *(const v8h*)(ap + 2 * arow + k + 8);
        a3.h[0] = *(const v8h*)(ap + 3 * arow + k);
        a3.h[1] = *(const v8h*)(ap + 3 * arow + k + 8);
        acc0 = __builtin_amdgcn_wmma_f32_16x16x32_f16(false, a0.v, false, b.v, (short)0, acc0, false, false);
        acc1 = __builtin_amdgcn_wmma_f32_16x16x32_f16(false, a1.v, false, b.v, (short)0, acc1, false, false);
        acc2 = __builtin_amdgcn_wmma_f32_16x16x32_f16(false, a2.v, false, b.v, (short)0, acc2, false, false);
        acc3 = __builtin_amdgcn_wmma_f32_16x16x32_f16(false, a3.v, false, b.v, (short)0, acc3, false, false);
    }
#endif

    const float bv = bias ? bias[col0 + mn] : 0.0f;
#pragma unroll
    for (int s = 0; s < 4; ++s) {
        const v8f acc = (s == 0) ? acc0 : (s == 1) ? acc1 : (s == 2) ? acc2 : acc3;
#pragma unroll
        for (int r = 0; r < 8; ++r) {
            const int m = (s << 4) + (hlf << 3) + r;
            float v = acc[r] + bv;
            if (relu) v = fmaxf(v, 0.0f);
            const size_t idx = (size_t)(row0 + m) * ldc + (col0 + mn);
            if (Cf) Cf[idx] = v;
            if (Ch) Ch[idx] = (f16)v;
        }
    }
}

// ---------------------------------------------------------------------------
// Weight preparation
// ---------------------------------------------------------------------------
__global__ void k_cvt_f16(const float* __restrict__ s, f16* __restrict__ d, int n)
{
    int i = blockIdx.x * 256 + threadIdx.x;
    if (i < n) d[i] = (f16)s[i];
}

// src is [K,N] row-major (x @ W convention); dst is [N,K] (K-contiguous).
__global__ void k_cvt_f16_T(const float* __restrict__ s, f16* __restrict__ d,
                            int K, int N)
{
    int i = blockIdx.x * 256 + threadIdx.x;
    if (i < K * N) {
        int k = i / N, n = i - k * N;
        d[(size_t)n * K + k] = (f16)s[i];
    }
}

__global__ void k_vadd(const float* __restrict__ a, const float* __restrict__ b,
                       float* __restrict__ o, int n)
{
    int i = blockIdx.x * 256 + threadIdx.x;
    if (i < n) o[i] = a[i] + b[i];
}

// ---------------------------------------------------------------------------
// Embedding MLP: [R,14] -> relu(14->48) -> (48->256). One block per row.
// ---------------------------------------------------------------------------
__global__ void k_embed(const float* __restrict__ x,
                        const float* __restrict__ w1, const float* __restrict__ b1,
                        const float* __restrict__ w2, const float* __restrict__ b2,
                        float* __restrict__ outf, f16* __restrict__ outh)
{
    const int r = blockIdx.x;
    const int e = threadIdx.x; // 256
    __shared__ float xin[cIN];
    __shared__ float h48[48];
    if (e < cIN) xin[e] = x[(size_t)r * cIN + e];
    __syncthreads();
    if (e < 48) {
        float s = b1[e];
#pragma unroll
        for (int i = 0; i < cIN; ++i) s += xin[i] * w1[i * 48 + e];
        h48[e] = fmaxf(s, 0.0f);
    }
    __syncthreads();
    float s = b2[e];
#pragma unroll 8
    for (int i = 0; i < 48; ++i) s += h48[i] * w2[i * cE + e];
    const size_t idx = (size_t)r * cE + e;
    outf[idx] = s;
    outh[idx] = (f16)s;
}

// ---------------------------------------------------------------------------
// Channel gate: fw1[b,e] = sum_t out1[b,t,e]*fw_w[t] + fw_b; softmax over e;
// zero the bottom-10 (stable-argsort tie behavior: smallest index wins ties).
// ---------------------------------------------------------------------------
__global__ void k_fw_gate(const float* __restrict__ out1,
                          const float* __restrict__ fww, const float* __restrict__ fwb,
                          float* __restrict__ Ag)
{
    const int b = blockIdx.x;
    const int e = threadIdx.x; // 256
    __shared__ float red[cE];
    __shared__ int   redi[cE];

    float s = fwb[0];
    const float* base = out1 + (size_t)b * cT * cE + e;
    for (int t = 0; t < cT; ++t) s += base[(size_t)t * cE] * fww[t];

    red[e] = s; __syncthreads();
    for (int st = 128; st > 0; st >>= 1) { if (e < st) red[e] = fmaxf(red[e], red[e + st]); __syncthreads(); }
    const float mx = red[0]; __syncthreads();
    const float ex = __expf(s - mx);
    red[e] = ex; __syncthreads();
    for (int st = 128; st > 0; st >>= 1) { if (e < st) red[e] += red[e + st]; __syncthreads(); }
    const float soft = ex / red[0];
    __syncthreads();

    float mask = 1.0f;
    for (int it = 0; it < 10; ++it) {
        red[e]  = (mask > 0.0f) ? soft : 3.0e38f;
        redi[e] = e;
        __syncthreads();
        for (int st = 128; st > 0; st >>= 1) {
            if (e < st) {
                float ov = red[e + st]; int oi = redi[e + st];
                if (ov < red[e] || (ov == red[e] && oi < redi[e])) { red[e] = ov; redi[e] = oi; }
            }
            __syncthreads();
        }
        if (e == redi[0]) mask = 0.0f;
        __syncthreads();
    }
    Ag[(size_t)b * cE + e] = soft * mask;
}

// ---------------------------------------------------------------------------
// LSTM pointwise cell: g = G[:,t,:] (+ Rt); torch gate order i,f,g,o.
// ---------------------------------------------------------------------------
__global__ void k_lstm_cell(const float* __restrict__ G, const float* __restrict__ Rt,
                            float* __restrict__ c, float* __restrict__ seqf,
                            f16* __restrict__ seqh, f16* __restrict__ hh, int t)
{
    const int b = blockIdx.x;
    const int e = threadIdx.x; // 256
    const size_t gb = ((size_t)b * cT + t) * cG;
    const size_t rb = (size_t)b * cG;
    float gi = G[gb + e];
    float gf = G[gb + cE + e];
    float gg = G[gb + 2 * cE + e];
    float go = G[gb + 3 * cE + e];
    if (t > 0) {
        gi += Rt[rb + e]; gf += Rt[rb + cE + e];
        gg += Rt[rb + 2 * cE + e]; go += Rt[rb + 3 * cE + e];
    }
    const float cp = (t > 0) ? c[(size_t)b * cE + e] : 0.0f;
    const float si = 1.0f / (1.0f + __expf(-gi));
    const float sf = 1.0f / (1.0f + __expf(-gf));
    const float so = 1.0f / (1.0f + __expf(-go));
    const float cn = sf * cp + si * tanhf(gg);
    const float hn = so * tanhf(cn);
    c[(size_t)b * cE + e] = cn;
    const size_t idx = ((size_t)b * cT + t) * cE + e;
    seqf[idx] = hn;
    seqh[idx] = (f16)hn;
    hh[(size_t)b * cE + e] = (f16)hn;
}

// ---------------------------------------------------------------------------
// LayerNorm over E=256 with optional residual (x2) and per-(b,e) gate.
// ---------------------------------------------------------------------------
__global__ void k_layernorm(const float* __restrict__ x1, const float* __restrict__ x2,
                            const float* __restrict__ gate,
                            const float* __restrict__ g, const float* __restrict__ bb,
                            float* __restrict__ yf, f16* __restrict__ yh)
{
    const int r = blockIdx.x;
    const int e = threadIdx.x; // 256
    const size_t idx = (size_t)r * cE + e;
    float h = x1[idx];
    if (x2) {
        float add = x2[idx];
        if (gate) add *= gate[(size_t)(r >> 6) * cE + e]; // b = r / T
        h += add;
    }
    __shared__ float red[cE];
    red[e] = h; __syncthreads();
    for (int st = 128; st > 0; st >>= 1) { if (e < st) red[e] += red[e + st]; __syncthreads(); }
    const float mean = red[0] * (1.0f / cE); __syncthreads();
    const float d = h - mean;
    red[e] = d * d; __syncthreads();
    for (int st = 128; st > 0; st >>= 1) { if (e < st) red[e] += red[e + st]; __syncthreads(); }
    const float inv = rsqrtf(red[0] * (1.0f / cE) + 1e-5f);
    const float y = d * inv * g[e] + bb[e];
    if (yf) yf[idx] = y;
    if (yh) yh[idx] = (f16)y;
}

// ---------------------------------------------------------------------------
// Attention scores: S[bh][tq][tk] = (q . k) * scale. K = dh = 32 -> one WMMA.
// ---------------------------------------------------------------------------
__global__ void k_attn_scores(const f16* __restrict__ qh, const f16* __restrict__ kh,
                              float* __restrict__ S, float scale)
{
    const int lane = threadIdx.x;
    const int bh = blockIdx.z, b = bh >> 3, hd = bh & 7;
    const int tq0 = blockIdx.x << 4, tk0 = blockIdx.y << 4;
    const int mn = lane & 15, hlf = lane >> 4;

    const f16* ap = qh + ((size_t)(b * cT + tq0 + mn) * cE) + hd * cDH + 16 * hlf;
    const f16* bp = kh + ((size_t)(b * cT + tk0 + mn) * cE) + hd * cDH + 16 * hlf;
    V16 a, bbv;
    a.h[0] = *(const v8h*)(ap);
    a.h[1] = *(const v8h*)(ap + 8);
    bbv.v  = *(const v16h*)(bp);
    v8f acc = {};
    acc = __builtin_amdgcn_wmma_f32_16x16x32_f16(false, a.v, false, bbv.v,
                                                 (short)0, acc, false, false);
#pragma unroll
    for (int r = 0; r < 8; ++r) {
        const int m = (hlf << 3) + r;
        S[((size_t)bh * cT + tq0 + m) * cT + tk0 + mn] = acc[r] * scale;
    }
}

// Row softmax over 64 keys; one block (64 threads = 2 waves) per (bh, tq).
__global__ void k_attn_softmax(const float* __restrict__ S, f16* __restrict__ P)
{
    const int row = blockIdx.x;
    const int t = threadIdx.x; // 64
    const float v = S[(size_t)row * cT + t];
    __shared__ float red[cT];
    red[t] = v; __syncthreads();
    for (int st = 32; st > 0; st >>= 1) { if (t < st) red[t] = fmaxf(red[t], red[t + st]); __syncthreads(); }
    const float mx = red[0]; __syncthreads();
    const float e = __expf(v - mx);
    red[t] = e; __syncthreads();
    for (int st = 32; st > 0; st >>= 1) { if (t < st) red[t] += red[t + st]; __syncthreads(); }
    P[(size_t)row * cT + t] = (f16)(e / red[0]);
}

// Transposed V: vT[bh][d][t] = v[b][t][hd*32+d]  (K-contiguous B operand)
__global__ void k_build_vT(const f16* __restrict__ vh, f16* __restrict__ vT)
{
    const size_t i = (size_t)blockIdx.x * 256 + threadIdx.x;
    const int t = (int)(i & 63);
    const size_t r = i >> 6;
    const int d = (int)(r & 31);
    const size_t bh = r >> 5;
    const int b = (int)(bh >> 3), hd = (int)(bh & 7);
    vT[i] = vh[((size_t)(b * cT + t)) * cE + hd * cDH + d];
}

// ctx = P @ V; K=64 (2 WMMA). grid (2,4,B*NH), block 32.
__global__ void k_attn_ctx(const f16* __restrict__ P, const f16* __restrict__ vT,
                           f16* __restrict__ ctx)
{
    const int lane = threadIdx.x;
    const int bh = blockIdx.z, b = bh >> 3, hd = bh & 7;
    const int tq0 = blockIdx.y << 4, d0 = blockIdx.x << 4;
    const int mn = lane & 15, hlf = lane >> 4;

    const f16* ap = P  + ((size_t)bh * cT + tq0 + mn) * cT + 16 * hlf;
    const f16* bp = vT + ((size_t)bh * cDH + d0 + mn) * cT + 16 * hlf;
    v8f acc = {};
    for (int k = 0; k < cT; k += 32) {
        V16 a, bbv;
        a.h[0] = *(const v8h*)(ap + k);
        a.h[1] = *(const v8h*)(ap + k + 8);
        bbv.v  = *(const v16h*)(bp + k);
        acc = __builtin_amdgcn_wmma_f32_16x16x32_f16(false, a.v, false, bbv.v,
                                                     (short)0, acc, false, false);
    }
#pragma unroll
    for (int r = 0; r < 8; ++r) {
        const int m = (hlf << 3) + r;
        ctx[((size_t)(b * cT + tq0 + m)) * cE + hd * cDH + d0 + mn] = (f16)acc[r];
    }
}

// ---------------------------------------------------------------------------
// Final N=1 projection: out[b] = hid[b,:] . cnn_w2 + cnn_b2
// ---------------------------------------------------------------------------
__global__ void k_final_out(const float* __restrict__ hid, const float* __restrict__ w2,
                            const float* __restrict__ b2, float* __restrict__ out)
{
    const int b = blockIdx.x;
    const int e = threadIdx.x; // 256
    float s = hid[(size_t)b * cLH + e] * w2[e] + hid[(size_t)b * cLH + e + 256] * w2[e + 256];
    __shared__ float red[256];
    red[e] = s; __syncthreads();
    for (int st = 128; st > 0; st >>= 1) { if (e < st) red[e] += red[e + st]; __syncthreads(); }
    if (e == 0) out[b] = red[0] + b2[0];
}

// ---------------------------------------------------------------------------
// Host orchestration
// ---------------------------------------------------------------------------
extern "C" void kernel_launch(void* const* d_in, const int* in_sizes, int n_in,
                              void* d_out, int out_size, void* d_ws, size_t ws_size,
                              hipStream_t stream)
{
    (void)in_sizes; (void)n_in; (void)out_size; (void)ws_size;

    const float* x      = (const float*)d_in[0];
    const float* emb_w1 = (const float*)d_in[1];
    const float* emb_b1 = (const float*)d_in[2];
    const float* emb_w2 = (const float*)d_in[3];
    const float* emb_b2 = (const float*)d_in[4];
    const float* fw_w   = (const float*)d_in[5];
    const float* fw_b   = (const float*)d_in[6];
    const float* wih[2] = { (const float*)d_in[7],  (const float*)d_in[11] };
    const float* whh[2] = { (const float*)d_in[8],  (const float*)d_in[12] };
    const float* bih[2] = { (const float*)d_in[9],  (const float*)d_in[13] };
    const float* bhh[2] = { (const float*)d_in[10], (const float*)d_in[14] };
    const float* ln_g   = (const float*)d_in[15];
    const float* ln_b   = (const float*)d_in[16];
    const float* wq = (const float*)d_in[17]; const float* bq = (const float*)d_in[18];
    const float* wk = (const float*)d_in[19]; const float* bk = (const float*)d_in[20];
    const float* wv = (const float*)d_in[21]; const float* bv = (const float*)d_in[22];
    const float* wo = (const float*)d_in[23]; const float* bo = (const float*)d_in[24];
    const float* eln1_g = (const float*)d_in[25]; const float* eln1_b = (const float*)d_in[26];
    const float* ff_w1 = (const float*)d_in[27]; const float* ff_b1 = (const float*)d_in[28];
    const float* ff_w2 = (const float*)d_in[29]; const float* ff_b2 = (const float*)d_in[30];
    const float* eln2_g = (const float*)d_in[31]; const float* eln2_b = (const float*)d_in[32];
    const float* cnn_w1 = (const float*)d_in[33]; const float* cnn_b1 = (const float*)d_in[34];
    const float* cnn_w2 = (const float*)d_in[35]; const float* cnn_b2 = (const float*)d_in[36];
    float* out = (float*)d_out;

    char* base = (char*)d_ws;
    size_t off = 0;
    auto alloc = [&](size_t bytes) -> void* {
        off = (off + 255) & ~(size_t)255;
        void* p = base + off;
        off += bytes;
        return p;
    };

    f16* wihH[2] = { (f16*)alloc((size_t)cG * cE * 2), (f16*)alloc((size_t)cG * cE * 2) };
    f16* whhH[2] = { (f16*)alloc((size_t)cG * cE * 2), (f16*)alloc((size_t)cG * cE * 2) };
    float* bsum[2] = { (float*)alloc(cG * 4), (float*)alloc(cG * 4) };
    f16* wqT = (f16*)alloc((size_t)cE * cE * 2);
    f16* wkT = (f16*)alloc((size_t)cE * cE * 2);
    f16* wvT = (f16*)alloc((size_t)cE * cE * 2);
    f16* woT = (f16*)alloc((size_t)cE * cE * 2);
    f16* ffw1T = (f16*)alloc((size_t)cLH * cE * 2);
    f16* ffw2T = (f16*)alloc((size_t)cE * cLH * 2);
    f16* cnnw1T = (f16*)alloc((size_t)cLH * (cT * cE) * 2);

    float* out1f = (float*)alloc((size_t)cR * cE * 4);
    f16*   out1h = (f16*)alloc((size_t)cR * cE * 2);
    float* Agate = (float*)alloc((size_t)cB * cE * 4);
    float* seqf  = (float*)alloc((size_t)cR * cE * 4);
    f16*   seqh  = (f16*)alloc((size_t)cR * cE * 2);
    float* Gbuf  = (float*)alloc((size_t)cR * cG * 4);
    float* Rtb   = (float*)alloc((size_t)cB * cG * 4);
    float* cbuf  = (float*)alloc((size_t)cB * cE * 4);
    f16*   hh    = (f16*)alloc((size_t)cB * cE * 2);
    float* out3f = (float*)alloc((size_t)cR * cE * 4);
    f16*   out3h = (f16*)alloc((size_t)cR * cE * 2);
    f16*   qh    = (f16*)alloc((size_t)cR * cE * 2);
    f16*   kh    = (f16*)alloc((size_t)cR * cE * 2);
    f16*   vh    = (f16*)alloc((size_t)cR * cE * 2);
    f16*   vT    = (f16*)alloc((size_t)cB * cNH * cDH * cT * 2);
    float* Sbuf  = (float*)alloc((size_t)cB * cNH * cT * cT * 4);
    f16*   Pbuf  = (f16*)alloc((size_t)cB * cNH * cT * cT * 2);
    f16*   ctxh  = (f16*)alloc((size_t)cR * cE * 2);
    float* attnf = (float*)alloc((size_t)cR * cE * 4);
    float* h1f   = (float*)alloc((size_t)cR * cE * 4);
    f16*   h1h   = (f16*)alloc((size_t)cR * cE * 2);
    f16*   ffhh  = (f16*)alloc((size_t)cR * cLH * 2);
    float* ffof  = (float*)alloc((size_t)cR * cE * 4);
    f16*   out4h = (f16*)alloc((size_t)cR * cE * 2);
    float* hidf  = (float*)alloc((size_t)cB * cLH * 4);

    auto gemm = [&](const f16* A, int lda, const f16* Bt, int ldb, const float* bias,
                    float* Cf, f16* Ch, int ldc, int M, int N, int K, int relu) {
        dim3 grid(N / 64, M / 64);
        gemm_wmma_f16<<<grid, dim3(128), 0, stream>>>(A, lda, Bt, ldb, bias, Cf, Ch, ldc, K, relu);
    };
    auto cvt = [&](const float* s, f16* d, int n) {
        k_cvt_f16<<<(n + 255) / 256, 256, 0, stream>>>(s, d, n);
    };
    auto cvtT = [&](const float* s, f16* d, int K, int N) {
        k_cvt_f16_T<<<(K * N + 255) / 256, 256, 0, stream>>>(s, d, K, N);
    };

    // ---- weight prep ----
    for (int l = 0; l < 2; ++l) {
        cvt(wih[l], wihH[l], cG * cE);
        cvt(whh[l], whhH[l], cG * cE);
        k_vadd<<<(cG + 255) / 256, 256, 0, stream>>>(bih[l], bhh[l], bsum[l], cG);
    }
    cvtT(wq, wqT, cE, cE);
    cvtT(wk, wkT, cE, cE);
    cvtT(wv, wvT, cE, cE);
    cvtT(wo, woT, cE, cE);
    cvtT(ff_w1, ffw1T, cE, cLH);
    cvtT(ff_w2, ffw2T, cLH, cE);
    cvtT(cnn_w1, cnnw1T, cT * cE, cLH);

    // ---- stage 1: embedding MLP ----
    k_embed<<<cR, cE, 0, stream>>>(x, emb_w1, emb_b1, emb_w2, emb_b2, out1f, out1h);

    // ---- stage 2: channel gate ----
    k_fw_gate<<<cB, cE, 0, stream>>>(out1f, fw_w, fw_b, Agate);

    // ---- stage 3: 2-layer LSTM ----
    for (int l = 0; l < 2; ++l) {
        const f16* Xh = (l == 0) ? out1h : seqh;
        gemm(Xh, cE, wihH[l], cE, bsum[l], Gbuf, nullptr, cG, cR, cG, cE, 0);
        for (int t = 0; t < cT; ++t) {
            if (t > 0)
                gemm(hh, cE, whhH[l], cE, nullptr, Rtb, nullptr, cG, cB, cG, cE, 0);
            k_lstm_cell<<<cB, cE, 0, stream>>>(Gbuf, Rtb, cbuf, seqf, seqh, hh, t);
        }
    }

    // ---- stage 4: out3 = LN(out2 + out1 * A) ----
    k_layernorm<<<cR, cE, 0, stream>>>(seqf, out1f, Agate, ln_g, ln_b, out3f, out3h);

    // ---- stage 5: MHA ----
    gemm(out3h, cE, wqT, cE, bq, nullptr, qh, cE, cR, cE, cE, 0);
    gemm(out3h, cE, wkT, cE, bk, nullptr, kh, cE, cR, cE, cE, 0);
    gemm(out3h, cE, wvT, cE, bv, nullptr, vh, cE, cR, cE, cE, 0);
    k_build_vT<<<(cB * cNH * cDH * cT) / 256, 256, 0, stream>>>(vh, vT);
    k_attn_scores<<<dim3(4, 4, cB * cNH), 32, 0, stream>>>(qh, kh, Sbuf, 0.17677669529663687f);
    k_attn_softmax<<<cB * cNH * cT, cT, 0, stream>>>(Sbuf, Pbuf);
    k_attn_ctx<<<dim3(2, 4, cB * cNH), 32, 0, stream>>>(Pbuf, vT, ctxh);
    gemm(ctxh, cE, woT, cE, bo, attnf, nullptr, cE, cR, cE, cE, 0);
    k_layernorm<<<cR, cE, 0, stream>>>(out3f, attnf, nullptr, eln1_g, eln1_b, h1f, h1h);

    // ---- stage 6: FFN ----
    gemm(h1h, cE, ffw1T, cE, ff_b1, nullptr, ffhh, cLH, cR, cLH, cE, 1);
    gemm(ffhh, cLH, ffw2T, cLH, ff_b2, ffof, nullptr, cE, cR, cE, cLH, 0);
    k_layernorm<<<cR, cE, 0, stream>>>(h1f, ffof, nullptr, eln2_g, eln2_b, nullptr, out4h);

    // ---- stage 7: final MLP over flattened [B, 16384] ----
    gemm(out4h, cT * cE, cnnw1T, cT * cE, cnn_b1, hidf, nullptr, cLH, cB, cLH, cT * cE, 1);
    k_final_out<<<cB, 256, 0, stream>>>(hidf, cnn_w2, cnn_b2, out);
}
